// MyOlmo3MLP_6665789243839
// MI455X (gfx1250) — matrix operand order repaired and checked by reading, not compiled, not run on hardware
//
#include <hip/hip_runtime.h>

// Problem dims (fixed by the reference)
#define HD 4096           // hidden
#define ID 11008          // intermediate
#define MD 8192           // B*S tokens

typedef _Float16 half_t;
typedef _Float16 v4h  __attribute__((ext_vector_type(4)));
typedef _Float16 v8h  __attribute__((ext_vector_type(8)));
typedef _Float16 v16h __attribute__((ext_vector_type(16)));
typedef float    v8f  __attribute__((ext_vector_type(8)));

// ---------------------------------------------------------------------------
// WMMA helper: D = A(16x32 f16) * B(32x16 f16) + C(16x16 f32)
// ---------------------------------------------------------------------------
__device__ __forceinline__ v8f wmma_f16(v16h a, v16h b, v8f c) {
    return __builtin_amdgcn_wmma_f32_16x16x32_f16(
        false, a, false, b, (short)0, c, false, false);
}

// Fragment-packed layout: each 16x32 tile = 1KB block of 512 halves.
// Lane l owns halves [l*16, l*16+16) of the block -> two adjacent 16B loads,
// fully coalesced across the wave (32 lanes x 32B = 1KB contiguous).
//  A-side block fill (per ISA A 16x32 layout):
//    lane = (row&15) + 16*((q>>3)&1),  halfidx = (q>>4)*8 + (q&7)
//  B-side block fill (per ISA B 32x16 layout, same mapping as A):
//    lane = (row&15) + 16*((q>>3)&1),  halfidx = (q>>4)*8 + (q&7)
__device__ __forceinline__ v16h load_frag_pk(const half_t* __restrict__ p) {
    v8h lo = *(const v8h*)(p);
    v8h hi = *(const v8h*)(p + 8);
    return __builtin_shufflevector(lo, hi,
        0, 1, 2, 3, 4, 5, 6, 7, 8, 9, 10, 11, 12, 13, 14, 15);
}

// ---------------------------------------------------------------------------
// Pre-pass: convert + fragment-pack. One-time scatter (~20-30us of HBM time).
// ---------------------------------------------------------------------------
// A-side pack: x fp32 [rows, K] -> packed f16 fragments
template <int K>
__global__ __launch_bounds__(256) void pack_a_kernel(const float* __restrict__ src,
                                                     half_t* __restrict__ dst,
                                                     long rows) {
    const long n4 = rows * (K / 4);
    long i = (long)blockIdx.x * blockDim.x + threadIdx.x;
    const long stride = (long)gridDim.x * blockDim.x;
    for (; i < n4; i += stride) {
        const int r  = (int)(i / (K / 4));
        const int q  = (int)(i - (long)r * (K / 4)) * 4;
        float4 v = ((const float4*)src)[i];
        const int qq   = q & 31;
        const int lane = (r & 15) + (((qq >> 3) & 1) << 4);
        const int hidx = ((qq >> 4) << 3) + (qq & 7);
        const long blk = (long)(r >> 4) * (K >> 5) + (q >> 5);
        v4h h;
        h[0] = (half_t)v.x; h[1] = (half_t)v.y;
        h[2] = (half_t)v.z; h[3] = (half_t)v.w;
        *(v4h*)(dst + blk * 512 + lane * 16 + hidx) = h;
    }
}

// B-side pack: int8-valued int32 weights [rows, K] -> packed f16 fragments
template <int K>
__global__ __launch_bounds__(256) void pack_b_kernel(const int* __restrict__ src,
                                                     half_t* __restrict__ dst,
                                                     long rows) {
    const long n4 = rows * (K / 4);
    long i = (long)blockIdx.x * blockDim.x + threadIdx.x;
    const long stride = (long)gridDim.x * blockDim.x;
    for (; i < n4; i += stride) {
        const int r  = (int)(i / (K / 4));
        const int q  = (int)(i - (long)r * (K / 4)) * 4;
        int4 v = ((const int4*)src)[i];
        const int qq   = q & 31;
        const int lane = (r & 15) + (((qq >> 3) & 1) << 4);
        const int hidx = ((qq >> 4) << 3) + (qq & 7);
        const long blk = (long)(r >> 4) * (K >> 5) + (q >> 5);
        v4h h;
        h[0] = (half_t)v.x; h[1] = (half_t)v.y;
        h[2] = (half_t)v.z; h[3] = (half_t)v.w;
        *(v4h*)(dst + blk * 512 + lane * 16 + hidx) = h;
    }
}

// ---------------------------------------------------------------------------
// Phase 1: fused gate+up GEMM, all-coalesced fragment loads.
// Grid (ID/128, MD/128); 8 waves = 2(M) x 4(N); wave tile 64(M) x 32(N) for
// BOTH g and u (A fragments shared). 16 WMMA : 16 b128 loads per K-step.
// Epilogue: fp32 dequant scales, g*u, LDS transpose D-layout -> A-fragment
// layout, coalesced packed store of t.
// ---------------------------------------------------------------------------
__global__ __launch_bounds__(256) void gate_up_kernel(
    const half_t* __restrict__ xp,   // packed A [MD/16][HD/32][512]
    const half_t* __restrict__ gp,   // packed B [ID/16][HD/32][512]
    const half_t* __restrict__ upk,  // packed B [ID/16][HD/32][512]
    const float*  __restrict__ gs,   // [ID]
    const float*  __restrict__ us,   // [ID]
    half_t*       __restrict__ tp)   // packed A [MD/16][ID/32][512]
{
    __shared__ half_t lds[8 * 64 * 32];   // 32 KB, private 4KB region per wave

    const int lane = threadIdx.x & 31;
    const int wave = threadIdx.x >> 5;
    const int wm = wave & 1;
    const int wn = wave >> 1;
    const int m0 = blockIdx.y * 128 + wm * 64;
    const int n0 = blockIdx.x * 128 + wn * 32;
    const int KB = HD / 32;              // 128 K-blocks

    const half_t* ax = xp  + ((size_t)(m0 >> 4) * KB) * 512 + lane * 16;
    const half_t* bg = gp  + ((size_t)(n0 >> 4) * KB) * 512 + lane * 16;
    const half_t* bu = upk + ((size_t)(n0 >> 4) * KB) * 512 + lane * 16;
    const size_t TS = (size_t)KB * 512;  // halves per row-tile of 16

    v8f accg[8], accu[8];
    const v8f vzero = 0.0f;
#pragma unroll
    for (int i = 0; i < 8; ++i) { accg[i] = vzero; accu[i] = vzero; }

    for (int kb = 0; kb < KB; ++kb) {
        const size_t ko = (size_t)kb * 512;
        v16h a0 = load_frag_pk(ax + ko);
        v16h a1 = load_frag_pk(ax + ko + TS);
        v16h a2 = load_frag_pk(ax + ko + 2 * TS);
        v16h a3 = load_frag_pk(ax + ko + 3 * TS);
        v16h g0 = load_frag_pk(bg + ko);
        v16h g1 = load_frag_pk(bg + ko + TS);
        v16h u0 = load_frag_pk(bu + ko);
        v16h u1 = load_frag_pk(bu + ko + TS);

        accg[0] = wmma_f16(a0, g0, accg[0]); accg[1] = wmma_f16(a0, g1, accg[1]);
        accg[2] = wmma_f16(a1, g0, accg[2]); accg[3] = wmma_f16(a1, g1, accg[3]);
        accg[4] = wmma_f16(a2, g0, accg[4]); accg[5] = wmma_f16(a2, g1, accg[5]);
        accg[6] = wmma_f16(a3, g0, accg[6]); accg[7] = wmma_f16(a3, g1, accg[7]);
        accu[0] = wmma_f16(a0, u0, accu[0]); accu[1] = wmma_f16(a0, u1, accu[1]);
        accu[2] = wmma_f16(a1, u0, accu[2]); accu[3] = wmma_f16(a1, u1, accu[3]);
        accu[4] = wmma_f16(a2, u0, accu[4]); accu[5] = wmma_f16(a2, u1, accu[5]);
        accu[6] = wmma_f16(a3, u0, accu[6]); accu[7] = wmma_f16(a3, u1, accu[7]);
    }

    // Epilogue: dequant, g*u, transpose through private LDS region.
    const int lr = lane & 15;
    const int lh = lane >> 4;
    const float sg0 = gs[n0 + lr],      su0 = us[n0 + lr];
    const float sg1 = gs[n0 + 16 + lr], su1 = us[n0 + 16 + lr];
    half_t* wl = lds + wave * (64 * 32);

    // D layout: vgpr r, lanes 0-15 -> M=r, lanes 16-31 -> M=r+8; N = lane&15.
#pragma unroll
    for (int tm = 0; tm < 4; ++tm) {
        v8f G0 = accg[tm * 2 + 0], U0 = accu[tm * 2 + 0];
        v8f G1 = accg[tm * 2 + 1], U1 = accu[tm * 2 + 1];
#pragma unroll
        for (int r = 0; r < 8; ++r) {
            const int row = tm * 16 + lh * 8 + r;
            wl[row * 32 + lr]      = (half_t)((G0[r] * sg0) * (U0[r] * su0));
            wl[row * 32 + 16 + lr] = (half_t)((G1[r] * sg1) * (U1[r] * su1));
        }
    }

    // Read back in A-fragment order (LDS in-order within wave) + packed store.
    const size_t TKB = ID / 32;          // 344 K-blocks of t
    half_t* tout = tp + ((size_t)(m0 >> 4) * TKB + (n0 >> 5)) * 512 + lane * 16;
#pragma unroll
    for (int tm = 0; tm < 4; ++tm) {
        const half_t* lp = wl + (tm * 16 + lr) * 32 + lh * 8;
        v8h lo = *(const v8h*)(lp);
        v8h hi = *(const v8h*)(lp + 16);
        *(v8h*)(tout + (size_t)tm * TKB * 512)     = lo;
        *(v8h*)(tout + (size_t)tm * TKB * 512 + 8) = hi;
    }
}

// ---------------------------------------------------------------------------
// Phase 2: down GEMM. Grid (HD/256, MD/128); 8 waves = 2(M) x 4(N);
// wave tile 64 x 64 -> 16 WMMA : 16 b128 loads per K-step.
// out[m,h] = ds[h] * sum_i t[m,i]*dw[h,i], fp32 row-major output.
// ---------------------------------------------------------------------------
__global__ __launch_bounds__(256) void down_kernel(
    const half_t* __restrict__ tp,   // packed A [MD/16][ID/32][512]
    const half_t* __restrict__ dp,   // packed B [HD/16][ID/32][512]
    const float*  __restrict__ dsc,  // [HD]
    float*        __restrict__ out)  // [MD, HD]
{
    const int lane = threadIdx.x & 31;
    const int wave = threadIdx.x >> 5;
    const int wm = wave & 1;
    const int wn = wave >> 1;
    const int m0 = blockIdx.y * 128 + wm * 64;
    const int n0 = blockIdx.x * 256 + wn * 64;
    const int KB = ID / 32;              // 344 K-blocks

    const half_t* ax = tp + ((size_t)(m0 >> 4) * KB) * 512 + lane * 16;
    const half_t* bx = dp + ((size_t)(n0 >> 4) * KB) * 512 + lane * 16;
    const size_t TS = (size_t)KB * 512;

    v8f acc[16];
    const v8f vzero = 0.0f;
#pragma unroll
    for (int i = 0; i < 16; ++i) acc[i] = vzero;

    for (int kb = 0; kb < KB; ++kb) {
        const size_t ko = (size_t)kb * 512;
        v16h a0 = load_frag_pk(ax + ko);
        v16h a1 = load_frag_pk(ax + ko + TS);
        v16h a2 = load_frag_pk(ax + ko + 2 * TS);
        v16h a3 = load_frag_pk(ax + ko + 3 * TS);
        v16h b0 = load_frag_pk(bx + ko);
        v16h b1 = load_frag_pk(bx + ko + TS);
        v16h b2 = load_frag_pk(bx + ko + 2 * TS);
        v16h b3 = load_frag_pk(bx + ko + 3 * TS);

        acc[0]  = wmma_f16(a0, b0, acc[0]);  acc[1]  = wmma_f16(a0, b1, acc[1]);
        acc[2]  = wmma_f16(a0, b2, acc[2]);  acc[3]  = wmma_f16(a0, b3, acc[3]);
        acc[4]  = wmma_f16(a1, b0, acc[4]);  acc[5]  = wmma_f16(a1, b1, acc[5]);
        acc[6]  = wmma_f16(a1, b2, acc[6]);  acc[7]  = wmma_f16(a1, b3, acc[7]);
        acc[8]  = wmma_f16(a2, b0, acc[8]);  acc[9]  = wmma_f16(a2, b1, acc[9]);
        acc[10] = wmma_f16(a2, b2, acc[10]); acc[11] = wmma_f16(a2, b3, acc[11]);
        acc[12] = wmma_f16(a3, b0, acc[12]); acc[13] = wmma_f16(a3, b1, acc[13]);
        acc[14] = wmma_f16(a3, b2, acc[14]); acc[15] = wmma_f16(a3, b3, acc[15]);
    }

    const int lr = lane & 15;
    const int lh = lane >> 4;
    float s[4];
#pragma unroll
    for (int tn = 0; tn < 4; ++tn) s[tn] = dsc[n0 + tn * 16 + lr];

#pragma unroll
    for (int tm = 0; tm < 4; ++tm) {
#pragma unroll
        for (int r = 0; r < 8; ++r) {
            const size_t row = (size_t)(m0 + tm * 16 + lh * 8 + r) * HD;
#pragma unroll
            for (int tn = 0; tn < 4; ++tn) {
                out[row + n0 + tn * 16 + lr] = acc[tm * 4 + tn][r] * s[tn];
            }
        }
    }
}

// ---------------------------------------------------------------------------
// Launch: inputs (x, gate_wq, gate_scale, up_wq, up_scale, down_wq, down_scale)
// ---------------------------------------------------------------------------
extern "C" void kernel_launch(void* const* d_in, const int* in_sizes, int n_in,
                              void* d_out, int out_size, void* d_ws, size_t ws_size,
                              hipStream_t stream) {
    (void)in_sizes; (void)n_in; (void)out_size; (void)ws_size;

    const float* x   = (const float*)d_in[0];
    const int*   gwq = (const int*)d_in[1];
    const float* gsc = (const float*)d_in[2];
    const int*   uwq = (const int*)d_in[3];
    const float* usc = (const float*)d_in[4];
    const int*   dwq = (const int*)d_in[5];
    const float* dsc = (const float*)d_in[6];
    float* out = (float*)d_out;

    // Workspace: xp 64MiB | gp 86MiB | up 86MiB | dp 86MiB | tp 172MiB
    char* ws = (char*)d_ws;
    const size_t XH_B = (size_t)MD * HD * sizeof(half_t);
    const size_t W_B  = (size_t)ID * HD * sizeof(half_t);
    half_t* xp = (half_t*)(ws);
    half_t* gp = (half_t*)(ws + XH_B);
    half_t* up = (half_t*)(ws + XH_B + W_B);
    half_t* dp = (half_t*)(ws + XH_B + 2 * W_B);
    half_t* tp = (half_t*)(ws + XH_B + 3 * W_B);

    // Pre-pass: convert + fragment-pack (stream-ordered).
    pack_a_kernel<HD><<<4096, 256, 0, stream>>>(x, xp, MD);
    pack_b_kernel<HD><<<4096, 256, 0, stream>>>(gwq, gp, ID);
    pack_b_kernel<HD><<<4096, 256, 0, stream>>>(uwq, up, ID);
    pack_b_kernel<ID><<<4096, 256, 0, stream>>>(dwq, dp, HD);

    // Fused gate/up GEMM -> t (fragment-packed f16)
    {
        dim3 grid(ID / 128, MD / 128);
        gate_up_kernel<<<grid, 256, 0, stream>>>(xp, gp, up, gsc, usc, tp);
    }

    // Down GEMM -> fp32 output
    {
        dim3 grid(HD / 256, MD / 128);
        down_kernel<<<grid, 256, 0, stream>>>(tp, dp, dsc, out);
    }
}
